// TemporalBipartiteGNN_89249420410964
// MI455X (gfx1250) — matrix operand order repaired and checked by reading, not compiled, not run on hardware
//
#include <hip/hip_runtime.h>
#include <hip/hip_bf16.h>
#include <cstdint>
#include <cstddef>

#define NCN 3000
#define NPN 30000
#define TSTEPS 8
#define NEDGE 64000
#define HD 128
#define NT 4            // 16x16 n-tiles per wave (16x64 output strip)
#define KMAX 128        // largest K used (all K in this net: 32/64/128)

typedef __attribute__((ext_vector_type(16))) __bf16 v16bf;
typedef __attribute__((ext_vector_type(8)))  __bf16 v8bf;
typedef __attribute__((ext_vector_type(8)))  float  v8f;

__device__ __forceinline__ __bf16 f2bf(float f) {
    unsigned int u = __builtin_bit_cast(unsigned int, f);
    u += 0x7FFFu + ((u >> 16) & 1u);           // round-to-nearest-even
    unsigned short s = (unsigned short)(u >> 16);
    return __builtin_bit_cast(__bf16, s);
}
__device__ __forceinline__ float bf2f(__bf16 b) {
    unsigned int u = (unsigned int)__builtin_bit_cast(unsigned short, b) << 16;
    return __builtin_bit_cast(float, u);
}

// ---------------------------------------------------------------------------
// W[O,K] f32 (reference: out = x @ W.T)  ->  transposed bf16 Wt[K,O]
// so a WMMA B-fragment row (fixed k, 16 consecutive n) is contiguous.
// ---------------------------------------------------------------------------
__global__ void wt_convert_kernel(const float* __restrict__ W,
                                  __bf16* __restrict__ Wt, int O, int K) {
    int idx = blockIdx.x * blockDim.x + threadIdx.x;
    if (idx >= O * K) return;
    int o = idx / K, k = idx - o * K;
    Wt[(size_t)k * O + o] = f2bf(W[(size_t)o * K + k]);
}

// flat f32 -> bf16 copy (for raw inputs)
__global__ void cvt_bf16_kernel(const float* __restrict__ src,
                                __bf16* __restrict__ dst, int n) {
    int idx = blockIdx.x * blockDim.x + threadIdx.x;
    if (idx < n) dst[idx] = f2bf(src[idx]);
}

// ---------------------------------------------------------------------------
// Fused GEMM: C[M,O] = act( A1@Wt1 + (HASA2 ? A2@Wt2 : 0) + bias )
// A: bf16 row-major [M,K];  Wt: bf16 [K,O];  K in {32,64,128}, O mult of 64.
// Grid: block = 8 consecutive 16-row m-tiles sharing ONE 64-column strip.
// The K x 64 weight strip(s) are staged once per block into LDS (<=32KB),
// so inner-loop B-fragments are 16-bit LDS reads at 128B k-stride (low latency) and weights are
// read from L2 once per block.  A-fragments stream from global (L2-resident).
// All addressing in the k-loop is strength-reduced to pointer bumps.
// HASA2 is a template param: no exec-masking branch ever surrounds WMMA.
// ---------------------------------------------------------------------------
template <bool HASA2>
__global__ __launch_bounds__(256) void gemm_bf16_wmma_kernel(
        const __bf16* __restrict__ A1, const __bf16* __restrict__ Wt1,
        const __bf16* __restrict__ A2, const __bf16* __restrict__ Wt2,
        const float* __restrict__ bias,
        float* __restrict__ Cf, __bf16* __restrict__ Cb,
        int M, int K, int O, int relu) {
    constexpr int NOPS = HASA2 ? 2 : 1;
    __shared__ __bf16 sW[NOPS * KMAX * 64];

    const int lane = threadIdx.x & 31;
    const int wave = threadIdx.x >> 5;
    const int ogroups = O >> 6;                  // strips of 64 columns
    const int mtiles = (M + 15) >> 4;
    const int bm = blockIdx.x / ogroups;
    const int og = blockIdx.x - bm * ogroups;
    const int n0 = og << 6;

    // ---- cooperative staging of the K x 64 strip(s): 128B/row = 8 x b128 ----
    {
        const int row0 = threadIdx.x >> 3;       // 32 rows per pass
        const int ch   = threadIdx.x & 7;        // 16B chunk in row
        for (int k = row0; k < K; k += 32) {
            ((uint4*)sW)[k * 8 + ch] =
                *(const uint4*)(Wt1 + (size_t)k * O + n0 + ch * 8);
            if constexpr (HASA2) {
                ((uint4*)(sW + KMAX * 64))[k * 8 + ch] =
                    *(const uint4*)(Wt2 + (size_t)k * O + n0 + ch * 8);
            }
        }
    }
    __syncthreads();

    const int mt = bm * 8 + wave;
    if (mt >= mtiles) return;                    // wave-uniform, after barrier
    const int m0 = mt << 4;
    const int hl = lane >> 4;                    // half: 0 or 1
    int ar = m0 + (lane & 15);
    if (ar >= M) ar = M - 1;                     // clamp (stores are guarded)

    // strength-reduced pointers (bumped per k-step)
    const __bf16* a1p = A1 + (size_t)ar * K + hl * 8;
    const __bf16* a2p = HASA2 ? (A2 + (size_t)ar * K + hl * 8) : nullptr;
    const __bf16* sB1 = sW + (lane & 15);        // column lane&15 of 64-col strip
    const __bf16* sB2 = sW + KMAX * 64 + (lane & 15);

    v8f acc[NT] = {};
    union AF { v16bf v; v8bf h[2]; };
#pragma unroll 2
    for (int k0 = 0; k0 < K; k0 += 32) {
        AF a1, a2;
        v16bf b1[NT], b2[NT];
        // ---- issue all loads for this k-step first ----
        a1.h[0] = *(const v8bf*)(a1p);
        a1.h[1] = *(const v8bf*)(a1p + 16);
#pragma unroll
        for (int nt = 0; nt < NT; ++nt)
            for (int s = 0; s < 16; ++s)
                b1[nt][s] = sB1[((s & 7) + hl * 8 + (s >> 3) * 16) * 64 + nt * 16];
        if constexpr (HASA2) {
            a2.h[0] = *(const v8bf*)(a2p);
            a2.h[1] = *(const v8bf*)(a2p + 16);
#pragma unroll
            for (int nt = 0; nt < NT; ++nt)
                for (int s = 0; s < 16; ++s)
                    b2[nt][s] = sB2[((s & 7) + hl * 8 + (s >> 3) * 16) * 64 + nt * 16];
        }
        __builtin_prefetch(a1p + 32, 0, 1);      // speculative next A slice
        // ---- WMMA burst ----
#pragma unroll
        for (int nt = 0; nt < NT; ++nt)
            acc[nt] = __builtin_amdgcn_wmma_f32_16x16x32_bf16(
                          false, a1.v, false, b1[nt], (short)0, acc[nt], false, false);
        if constexpr (HASA2) {
#pragma unroll
            for (int nt = 0; nt < NT; ++nt)
                acc[nt] = __builtin_amdgcn_wmma_f32_16x16x32_bf16(
                              false, a2.v, false, b2[nt], (short)0, acc[nt], false, false);
        }
        a1p += 32;
        if constexpr (HASA2) a2p += 32;
        sB1 += 32 * 64;
        sB2 += 32 * 64;
    }
    // C/D layout: VGPR r -> row m0+r+half*8, lane&15 -> column
#pragma unroll
    for (int nt = 0; nt < NT; ++nt) {
        const int col = n0 + nt * 16 + (lane & 15);
        const float bv = bias ? bias[col] : 0.0f;
#pragma unroll
        for (int r = 0; r < 8; ++r) {
            int row = m0 + r + hl * 8;
            if (row < M) {
                float v = acc[nt][r] + bv;
                if (relu) v = fmaxf(v, 0.0f);
                if (Cf) Cf[(size_t)row * O + col] = v;
                if (Cb) Cb[(size_t)row * O + col] = f2bf(v);
            }
        }
    }
}

// ---------------------------------------------------------------------------
// Scatter-add for segment mean: accum[dst[e]] += feat[src[e]], cnt[dst[e]]++
// feat is bf16 (activations); accumulation in f32 atomics for precision.
// One thread per (edge, 4 features).
// ---------------------------------------------------------------------------
__global__ void scatter_add_kernel(const __bf16* __restrict__ feat,
                                   const int* __restrict__ src,
                                   const int* __restrict__ dst,
                                   float* __restrict__ accum,
                                   float* __restrict__ cnt, int E) {
    const int per_edge = HD / 4;
    int idx = blockIdx.x * blockDim.x + threadIdx.x;
    if (idx >= E * per_edge) return;
    int e  = idx / per_edge;
    int f4 = (idx - e * per_edge) * 4;
    int s = src[e], d = dst[e];
    const __bf16* fp = feat + (size_t)s * HD + f4;
    float* ap = accum + (size_t)d * HD + f4;
    atomicAdd(ap + 0, bf2f(fp[0]));
    atomicAdd(ap + 1, bf2f(fp[1]));
    atomicAdd(ap + 2, bf2f(fp[2]));
    atomicAdd(ap + 3, bf2f(fp[3]));
    if (f4 == 0) atomicAdd(cnt + d, 1.0f);
}

// mean = accum / max(cnt,1), emitted as bf16 for the consuming GEMM
__global__ void seg_div_kernel(const float* __restrict__ accum,
                               const float* __restrict__ cnt,
                               __bf16* __restrict__ outb, int N) {
    int idx = blockIdx.x * blockDim.x + threadIdx.x;
    if (idx >= N * HD) return;
    outb[idx] = f2bf(accum[idx] / fmaxf(cnt[idx / HD], 1.0f));
}

// ---------------------------------------------------------------------------
// GRU gates (torch ordering r,z,n): h2 = (1-z)*tanh(in + r*hn) + z*h
// Writes h2 as f32 (recurrence + final output) and bf16 (next h@W_hh GEMM).
// ---------------------------------------------------------------------------
__global__ void gru_gate_kernel(const float* __restrict__ gi,
                                const float* __restrict__ gh,
                                const float* __restrict__ h,
                                float* __restrict__ h2f,
                                __bf16* __restrict__ h2b, int N) {
    int idx = blockIdx.x * blockDim.x + threadIdx.x;
    if (idx >= N * HD) return;
    int n = idx / HD, j = idx - n * HD;
    const float* gin = gi + (size_t)n * 3 * HD;
    const float* ghn = gh + (size_t)n * 3 * HD;
    float r  = 1.0f / (1.0f + expf(-(gin[j] + ghn[j])));
    float z  = 1.0f / (1.0f + expf(-(gin[HD + j] + ghn[HD + j])));
    float nn = tanhf(gin[2 * HD + j] + r * ghn[2 * HD + j]);
    float v = (1.0f - z) * nn + z * h[idx];
    h2f[idx] = v;
    h2b[idx] = f2bf(v);
}

// ---------------------------------------------------------------------------
extern "C" void kernel_launch(void* const* d_in, const int* in_sizes, int n_in,
                              void* d_out, int out_size, void* d_ws, size_t ws_size,
                              hipStream_t stream) {
    (void)in_sizes; (void)n_in; (void)out_size; (void)ws_size;
    const float* x_country = (const float*)d_in[0];   // [T,NC,32]
    const float* x_product = (const float*)d_in[1];   // [T,NP,64]
    const int*   edge_src  = (const int*)  d_in[2];   // [T,E]
    const int*   edge_dst  = (const int*)  d_in[3];   // [T,E]
    const float* w_clin = (const float*)d_in[4];
    const float* b_clin = (const float*)d_in[5];
    const float* w_plin = (const float*)d_in[6];
    const float* b_plin = (const float*)d_in[7];
    const float* w_sage[8]; for (int i = 0; i < 8; ++i) w_sage[i] = (const float*)d_in[8 + i];
    // order: wl1_cp, wr1_cp, wl1_pc, wr1_pc, wl2_cp, wr2_cp, wl2_pc, wr2_pc
    const float* bl1_cp = (const float*)d_in[16];
    const float* bl1_pc = (const float*)d_in[17];
    const float* bl2_cp = (const float*)d_in[18];
    const float* bl2_pc = (const float*)d_in[19];
    const float* w_ih_c = (const float*)d_in[20];
    const float* w_hh_c = (const float*)d_in[21];
    const float* w_ih_p = (const float*)d_in[22];
    const float* w_hh_p = (const float*)d_in[23];
    const float* b_ih_c = (const float*)d_in[24];
    const float* b_hh_c = (const float*)d_in[25];
    const float* b_ih_p = (const float*)d_in[26];
    const float* b_hh_p = (const float*)d_in[27];

    char* ws = (char*)d_ws;
    size_t off = 0;
    auto alloc = [&](size_t bytes) -> void* {
        size_t o = (off + 255) & ~(size_t)255;
        off = o + bytes;
        return (void*)(ws + o);
    };

    // bf16 transposed weights
    __bf16* wt_clin = (__bf16*)alloc((size_t)32 * HD * 2);
    __bf16* wt_plin = (__bf16*)alloc((size_t)64 * HD * 2);
    __bf16* wt_sage[8];
    for (int i = 0; i < 8; ++i) wt_sage[i] = (__bf16*)alloc((size_t)HD * HD * 2);
    __bf16* wt_gru[4];
    for (int i = 0; i < 4; ++i) wt_gru[i] = (__bf16*)alloc((size_t)HD * 3 * HD * 2);

    // bf16 copies of raw inputs (converted once, used all T steps)
    __bf16* xc_bf = (__bf16*)alloc((size_t)TSTEPS * NCN * 32 * 2);
    __bf16* xp_bf = (__bf16*)alloc((size_t)TSTEPS * NPN * 64 * 2);

    // per-step embeddings (bf16: only ever consumed as GEMM A operands)
    __bf16* zc_seq = (__bf16*)alloc((size_t)TSTEPS * NCN * HD * 2);
    __bf16* zp_seq = (__bf16*)alloc((size_t)TSTEPS * NPN * HD * 2);

    // scratch arena (time-shared between encode phase and GRU phase)
    const size_t enc_bytes =
        ((size_t)NCN * HD * 2) * 3 +            // hc_bf, hc1_bf, mc_bf
        ((size_t)NPN * HD * 2) * 3 +            // hp_bf, hp1_bf, mp_bf
        (size_t)NCN * HD * 4 + (size_t)NPN * HD * 4 +  // f32 accumulators
        (size_t)NCN * 4 + (size_t)NPN * 4 + 4096;      // counts
    const size_t gru_bytes =
        2 * (size_t)NPN * 3 * HD * 4 +          // gi, gh
        2 * (size_t)NPN * HD * 4 +              // hA, hB (f32)
        (size_t)NPN * HD * 2 + 4096;            // h_bf
    char* arena = (char*)alloc(enc_bytes > gru_bytes ? enc_bytes : gru_bytes);

    // encode-phase overlay
    __bf16* hc_bf  = (__bf16*)arena;
    __bf16* hc1_bf = hc_bf  + (size_t)NCN * HD;
    __bf16* mc_bf  = hc1_bf + (size_t)NCN * HD;
    __bf16* hp_bf  = mc_bf  + (size_t)NCN * HD;
    __bf16* hp1_bf = hp_bf  + (size_t)NPN * HD;
    __bf16* mp_bf  = hp1_bf + (size_t)NPN * HD;
    float*  mc_f   = (float*)(mp_bf + (size_t)NPN * HD);
    float*  mp_f   = mc_f + (size_t)NCN * HD;
    float*  cnt_c  = mp_f + (size_t)NPN * HD;
    float*  cnt_p  = cnt_c + NCN;

    auto gemm1 = [&](const __bf16* A1, const __bf16* W1, const float* bias,
                     float* Cf, __bf16* Cb, int M, int K, int O) {
        int blocks = (((M + 15) / 16 + 7) / 8) * (O / 64);
        gemm_bf16_wmma_kernel<false><<<blocks, 256, 0, stream>>>(
            A1, W1, nullptr, nullptr, bias, Cf, Cb, M, K, O, 0);
    };
    auto gemm2 = [&](const __bf16* A1, const __bf16* W1, const __bf16* A2,
                     const __bf16* W2, const float* bias, __bf16* Cb,
                     int M, int K, int O, int relu) {
        int blocks = (((M + 15) / 16 + 7) / 8) * (O / 64);
        gemm_bf16_wmma_kernel<true><<<blocks, 256, 0, stream>>>(
            A1, W1, A2, W2, bias, nullptr, Cb, M, K, O, relu);
    };
    auto wconv = [&](const float* W, __bf16* Wt, int O, int K) {
        int n = O * K;
        wt_convert_kernel<<<(n + 255) / 256, 256, 0, stream>>>(W, Wt, O, K);
    };

    // ---- one-time conversions (redone every call for determinism) ----
    wconv(w_clin, wt_clin, HD, 32);
    wconv(w_plin, wt_plin, HD, 64);
    for (int i = 0; i < 8; ++i) wconv(w_sage[i], wt_sage[i], HD, HD);
    wconv(w_ih_c, wt_gru[0], 3 * HD, HD);
    wconv(w_hh_c, wt_gru[1], 3 * HD, HD);
    wconv(w_ih_p, wt_gru[2], 3 * HD, HD);
    wconv(w_hh_p, wt_gru[3], 3 * HD, HD);
    {
        int n1 = TSTEPS * NCN * 32;
        cvt_bf16_kernel<<<(n1 + 255) / 256, 256, 0, stream>>>(x_country, xc_bf, n1);
        int n2 = TSTEPS * NPN * 64;
        cvt_bf16_kernel<<<(n2 + 255) / 256, 256, 0, stream>>>(x_product, xp_bf, n2);
    }

    const int sc_blk = (NEDGE * (HD / 4) + 255) / 256;
    const int dvP = (NPN * HD + 255) / 256;
    const int dvC = (NCN * HD + 255) / 256;

    // ---- per-step encoder ----
    for (int t = 0; t < TSTEPS; ++t) {
        const __bf16* xc = xc_bf + (size_t)t * NCN * 32;
        const __bf16* xp = xp_bf + (size_t)t * NPN * 64;
        const int* es = edge_src + (size_t)t * NEDGE;
        const int* ed = edge_dst + (size_t)t * NEDGE;

        gemm1(xc, wt_clin, b_clin, nullptr, hc_bf, NCN, 32, HD);
        gemm1(xp, wt_plin, b_plin, nullptr, hp_bf, NPN, 64, HD);

        // layer 1 messages
        hipMemsetAsync(mp_f, 0, (size_t)NPN * HD * 4, stream);
        hipMemsetAsync(mc_f, 0, (size_t)NCN * HD * 4, stream);
        hipMemsetAsync(cnt_p, 0, (size_t)NPN * 4, stream);
        hipMemsetAsync(cnt_c, 0, (size_t)NCN * 4, stream);
        scatter_add_kernel<<<sc_blk, 256, 0, stream>>>(hc_bf, es, ed, mp_f, cnt_p, NEDGE);
        scatter_add_kernel<<<sc_blk, 256, 0, stream>>>(hp_bf, ed, es, mc_f, cnt_c, NEDGE);
        seg_div_kernel<<<dvP, 256, 0, stream>>>(mp_f, cnt_p, mp_bf, NPN);
        seg_div_kernel<<<dvC, 256, 0, stream>>>(mc_f, cnt_c, mc_bf, NCN);

        gemm2(mp_bf, wt_sage[0], hp_bf, wt_sage[1], bl1_cp, hp1_bf, NPN, HD, HD, 1);
        gemm2(mc_bf, wt_sage[2], hc_bf, wt_sage[3], bl1_pc, hc1_bf, NCN, HD, HD, 1);

        // layer 2 messages
        hipMemsetAsync(mp_f, 0, (size_t)NPN * HD * 4, stream);
        hipMemsetAsync(mc_f, 0, (size_t)NCN * HD * 4, stream);
        hipMemsetAsync(cnt_p, 0, (size_t)NPN * 4, stream);
        hipMemsetAsync(cnt_c, 0, (size_t)NCN * 4, stream);
        scatter_add_kernel<<<sc_blk, 256, 0, stream>>>(hc1_bf, es, ed, mp_f, cnt_p, NEDGE);
        scatter_add_kernel<<<sc_blk, 256, 0, stream>>>(hp1_bf, ed, es, mc_f, cnt_c, NEDGE);
        seg_div_kernel<<<dvP, 256, 0, stream>>>(mp_f, cnt_p, mp_bf, NPN);
        seg_div_kernel<<<dvC, 256, 0, stream>>>(mc_f, cnt_c, mc_bf, NCN);

        gemm2(mp_bf, wt_sage[4], hp1_bf, wt_sage[5], bl2_cp,
              zp_seq + (size_t)t * NPN * HD, NPN, HD, HD, 0);
        gemm2(mc_bf, wt_sage[6], hc1_bf, wt_sage[7], bl2_pc,
              zc_seq + (size_t)t * NCN * HD, NCN, HD, HD, 0);
    }

    // ---- GRU over T (arena reused: encode scratch is dead now) ----
    auto run_gru = [&](const __bf16* seq, int N, const __bf16* wih,
                       const __bf16* whh, const float* bih, const float* bhh,
                       float* outp) {
        float* gi = (float*)arena;
        float* gh = gi + (size_t)NPN * 3 * HD;
        float* hA = gh + (size_t)NPN * 3 * HD;
        float* hB = hA + (size_t)NPN * HD;
        __bf16* h_bf = (__bf16*)(hB + (size_t)NPN * HD);
        hipMemsetAsync(hA, 0, (size_t)N * HD * 4, stream);
        hipMemsetAsync(h_bf, 0, (size_t)N * HD * 2, stream);  // bf16 zero == 0x0000
        float* hcur = hA; float* hnext = hB;
        int gb = (N * HD + 255) / 256;
        for (int t = 0; t < TSTEPS; ++t) {
            gemm1(seq + (size_t)t * N * HD, wih, bih, gi, nullptr, N, HD, 3 * HD);
            gemm1(h_bf, whh, bhh, gh, nullptr, N, HD, 3 * HD);
            gru_gate_kernel<<<gb, 256, 0, stream>>>(gi, gh, hcur, hnext, h_bf, N);
            float* tmp = hcur; hcur = hnext; hnext = tmp;
        }
        hipMemcpyAsync(outp, hcur, (size_t)N * HD * 4,
                       hipMemcpyDeviceToDevice, stream);
    };

    float* out = (float*)d_out;
    run_gru(zc_seq, NCN, wt_gru[0], wt_gru[1], b_ih_c, b_hh_c, out);
    run_gru(zp_seq, NPN, wt_gru[2], wt_gru[3], b_ih_p, b_hh_p, out + (size_t)NCN * HD);
}
